// ARDecoder_51101520887867
// MI455X (gfx1250) — matrix operand order repaired and checked by reading, not compiled, not run on hardware
//
#include <hip/hip_runtime.h>
#include <hip/hip_bf16.h>
#include <stdint.h>

// ---------------------------------------------------------------------------
// Autoregressive GRU decoder for MI455X (gfx1250, wave32, WMMA).
//
//   N=16384 samples, H=256 hidden, 24 steps.
//   gi collapses to x*u + c (u,c precomputed), so per step the only GEMM is
//   gh = h @ W_hh^T, done with v_wmma_f32_16x16x32_bf16 (f32 accumulate).
//   Samples are independent -> one persistent kernel runs all 24 steps with
//   h resident in LDS (bf16, double buffered).
//
//   Round-4: (a) first K-slice WMMAs use inline-0 C operand (no 96-mov acc
//   zero init); (b) gate nonlinearities use the hardware V_TANH_F32 trans op
//   (sigmoid(x) = 0.5*tanh(x/2)+0.5), fast-rcp fallback otherwise.
// ---------------------------------------------------------------------------

#define HID       256
#define NSAMP     16384
#define PRED_LEN  24
#define ROWS      64      // samples per workgroup
#define BLOCK     256     // 8 wave32 waves

typedef __attribute__((ext_vector_type(16))) __bf16 v16bf;
typedef __attribute__((ext_vector_type(8)))  float  v8f;

struct B16x16 { uint4 lo; uint4 hi; };   // 16 bf16 = 32 bytes
struct BTri   { v16bf r, z, n; };        // one K-slice of B for all 3 gates

__device__ __forceinline__ uint16_t f32_to_bf16(float f) {
  uint32_t x = __builtin_bit_cast(uint32_t, f);
  x += 0x7FFFu + ((x >> 16) & 1u);       // round-to-nearest-even
  return (uint16_t)(x >> 16);
}
__device__ __forceinline__ float bf16_to_f32(uint16_t h) {
  uint32_t x = ((uint32_t)h) << 16;
  return __builtin_bit_cast(float, x);
}

// Build a v16bf fragment from two 16-byte aligned groups of 8 bf16.
__device__ __forceinline__ v16bf load_frag(const uint16_t* p0, const uint16_t* p1) {
  B16x16 t;
  t.lo = *(const uint4*)p0;
  t.hi = *(const uint4*)p1;
  return __builtin_bit_cast(v16bf, t);
}

// Fast tanh: hardware V_TANH_F32 on gfx1250 if the builtin exists.
__device__ __forceinline__ float fast_tanh(float x) {
#if __has_builtin(__builtin_amdgcn_tanhf)
  return __builtin_amdgcn_tanhf(x);
#elif __has_builtin(__builtin_amdgcn_tanh_f32)
  return __builtin_amdgcn_tanh_f32(x);
#else
  // tanh(x) = 1 - 2/(e^{2x}+1), fast rcp
  float e = __expf(2.0f * x);
  return 1.0f - 2.0f * __builtin_amdgcn_rcpf(e + 1.0f);
#endif
}
// sigmoid(x) = 0.5*tanh(x/2) + 0.5  (one trans op + 2 FMA)
__device__ __forceinline__ float sigm(float x) {
  return fmaf(0.5f, fast_tanh(0.5f * x), 0.5f);
}

// ---------------------------------------------------------------------------
// Prologue 1: u = W_ih @ w_proj, c = W_ih @ b_proj + b_ih   (768 dots of 256)
// ---------------------------------------------------------------------------
__global__ void gi_precompute_kernel(const float* __restrict__ W_ih,
                                     const float* __restrict__ b_ih,
                                     const float* __restrict__ w_proj,
                                     const float* __restrict__ b_proj,
                                     float* __restrict__ u,
                                     float* __restrict__ c) {
  int j = blockIdx.x * blockDim.x + threadIdx.x;
  if (j >= 3 * HID) return;
  const float* wr = W_ih + (size_t)j * HID;
  float su = 0.f, sc = 0.f;
  for (int k = 0; k < HID; ++k) {
    float w = wr[k];
    su = fmaf(w, w_proj[k], su);
    sc = fmaf(w, b_proj[k], sc);
  }
  u[j] = su;
  c[j] = sc + b_ih[j];
}

// ---------------------------------------------------------------------------
// Prologue 2: W_hh f32 [768,256] -> bf16 row-major copy in workspace
// ---------------------------------------------------------------------------
__global__ void whh_bf16_kernel(const float* __restrict__ W_hh,
                                uint16_t* __restrict__ Wb) {
  int i = blockIdx.x * blockDim.x + threadIdx.x;   // 0 .. 196607
  Wb[i] = f32_to_bf16(W_hh[i]);
}

// ---------------------------------------------------------------------------
// Main persistent decode kernel: one workgroup = 64 samples, all 24 steps.
//   wave w owns hidden-unit column tiles jt = 32w and 32w+16 (16 units each),
//   across all 4 row tiles of 16 samples.
// ---------------------------------------------------------------------------
__global__ __launch_bounds__(BLOCK)
void decode_kernel(const float* __restrict__ enc,
                   const uint16_t* __restrict__ Wb,   // bf16 W_hh [768,256]
                   const float* __restrict__ u,       // [768]
                   const float* __restrict__ c,       // [768]
                   const float* __restrict__ b_hh,    // [768]
                   const float* __restrict__ w_out,   // [256]
                   const float* __restrict__ b_out,   // [1]
                   float* __restrict__ out) {         // [N,24]
  __shared__ uint16_t hbuf[2][ROWS * HID];  // bf16 hidden state, double buffer
  __shared__ float    xs[ROWS];             // per-sample scalar step input
  __shared__ float    predsum[ROWS];        // h_new . w_out accumulator

  const int tid  = threadIdx.x;
  const int lane = tid & 31;
  const int wave = tid >> 5;
  const int hi   = lane >> 4;       // 16-lane half: 0 or 1
  const int ln   = lane & 15;
  const int rowbase = blockIdx.x * ROWS;

  // --- init: h0 = encoder_out tile (f32 -> bf16), x0 = 0, predsum = 0
  for (int i = tid; i < ROWS * HID; i += BLOCK)
    hbuf[0][i] = f32_to_bf16(enc[(size_t)rowbase * HID + i]);
  if (tid < ROWS) { xs[tid] = 0.f; predsum[tid] = 0.f; }
  __syncthreads();

  const float bout = b_out[0];

  for (int t = 0; t < PRED_LEN; ++t) {
    const uint16_t* hcur  = hbuf[t & 1];
    uint16_t*       hnext = hbuf[(t + 1) & 1];

    #pragma unroll 1
    for (int jj = 0; jj < 2; ++jj) {
      const int jt = wave * 32 + jj * 16;   // this wave's hidden-unit tile
      const int j  = jt + ln;               // this lane's hidden unit (B col)

      // gate parameters for this lane's hidden unit
      const float u_r = u[j],           c_r = c[j],           bh_r = b_hh[j];
      const float u_z = u[HID + j],     c_z = c[HID + j],     bh_z = b_hh[HID + j];
      const float u_n = u[2*HID + j],   c_n = c[2*HID + j],   bh_n = b_hh[2*HID + j];
      const float wo  = w_out[j];

      v8f acc[4][3];   // defined by the first (C=0) K-slice below

      // B fragment base: column j = W_hh row (g*256 + j); K(e) = hi*16 + e
      const uint16_t* brow_r = Wb + ((size_t)(0*HID + j) * HID + hi*16);
      const uint16_t* brow_z = Wb + ((size_t)(1*HID + j) * HID + hi*16);
      const uint16_t* brow_n = Wb + ((size_t)(2*HID + j) * HID + hi*16);

      auto loadB = [&](int kb) {
        BTri b;
        b.r = load_frag(brow_r + kb, brow_r + kb + 8);
        b.z = load_frag(brow_z + kb, brow_z + kb + 8);
        b.n = load_frag(brow_n + kb, brow_n + kb + 8);
        return b;
      };
      auto loadA = [&](v16bf A[4], int kb) {
        #pragma unroll
        for (int mt = 0; mt < 4; ++mt) {
          // A fragment: row m = mt*16 + ln;
          // K(e) = kb + hi*16 + e  -> two 16B groups
          const uint16_t* a0 = hcur + ((mt*16 + ln) * HID + kb + hi*16);
          A[mt] = load_frag(a0, a0 + 8);
        }
      };
      // 12 WMMAs for one K-slice, accumulating into acc.
      auto mmac = [&](const BTri& B, int kb) {
        v16bf A[4];
        loadA(A, kb);
        #pragma unroll
        for (int mt = 0; mt < 4; ++mt) {
          acc[mt][0] = __builtin_amdgcn_wmma_f32_16x16x32_bf16(
              false, A[mt], false, B.r, (short)0, acc[mt][0], false, false);
          acc[mt][1] = __builtin_amdgcn_wmma_f32_16x16x32_bf16(
              false, A[mt], false, B.z, (short)0, acc[mt][1], false, false);
          acc[mt][2] = __builtin_amdgcn_wmma_f32_16x16x32_bf16(
              false, A[mt], false, B.n, (short)0, acc[mt][2], false, false);
        }
      };
      // First K-slice: C is the literal zero vector -> inline 0 operand,
      // acc is *defined* here (no separate 96-register zero init).
      auto mmac0 = [&](const BTri& B) {
        const v8f zz = (v8f){0.f,0.f,0.f,0.f,0.f,0.f,0.f,0.f};
        v16bf A[4];
        loadA(A, 0);
        #pragma unroll
        for (int mt = 0; mt < 4; ++mt) {
          acc[mt][0] = __builtin_amdgcn_wmma_f32_16x16x32_bf16(
              false, A[mt], false, B.r, (short)0, zz, false, false);
          acc[mt][1] = __builtin_amdgcn_wmma_f32_16x16x32_bf16(
              false, A[mt], false, B.z, (short)0, zz, false, false);
          acc[mt][2] = __builtin_amdgcn_wmma_f32_16x16x32_bf16(
              false, A[mt], false, B.n, (short)0, zz, false, false);
        }
      };

      // --- GEMM: gh(:, jt tiles of all 3 gates) = h @ W_hh^T, K = 256
      // Software pipeline, even/odd B buffers (SSA-friendly, no reg copies).
      BTri B0 = loadB(0);
      BTri B1 = loadB(32);
      mmac0(B0);                 // ks = 0, C = 0
      B0 = loadB(64);
      mmac(B1, 32);              // ks = 1
      #pragma unroll 1
      for (int ks = 2; ks < 8; ks += 2) {
        B1 = loadB((ks + 1) * 32);
        mmac(B0, ks * 32);
        if (ks + 2 < 8) B0 = loadB((ks + 2) * 32);
        mmac(B1, (ks + 1) * 32);
      }

      // --- fused GRU gate math + h update + output-dot partials
      // D layout: VGPR r, lanes 0-15 -> (M=r, N=ln); lanes 16-31 -> (M=r+8, N=ln)
      #pragma unroll
      for (int mt = 0; mt < 4; ++mt) {
        #pragma unroll
        for (int rI = 0; rI < 8; ++rI) {
          const int m  = mt*16 + rI + hi*8;         // sample row in tile
          const float xv = xs[m];
          const float hv = bf16_to_f32(hcur[m * HID + j]);
          const float gr = acc[mt][0][rI] + bh_r + xv * u_r + c_r;
          const float gz = acc[mt][1][rI] + bh_z + xv * u_z + c_z;
          const float r  = sigm(gr);
          const float z  = sigm(gz);
          const float gn = xv * u_n + c_n + r * (acc[mt][2][rI] + bh_n);
          const float n  = fast_tanh(gn);
          const float hn = (1.0f - z) * n + z * hv;
          hnext[m * HID + j] = f32_to_bf16(hn);

          // partial of pred[m] = sum_j h_new[m,j]*w_out[j] over this 16-lane group
          float p = hn * wo;
          p += __shfl_xor(p, 1);
          p += __shfl_xor(p, 2);
          p += __shfl_xor(p, 4);
          p += __shfl_xor(p, 8);
          if (ln == 0) atomicAdd(&predsum[m], p);
        }
      }
    }
    __syncthreads();

    // --- finalize prediction, feed back as next x
    if (tid < ROWS) {
      const float p = predsum[tid] + bout;
      out[(size_t)(rowbase + tid) * PRED_LEN + t] = p;
      xs[tid] = p;
      predsum[tid] = 0.f;
    }
    __syncthreads();
  }
}

// ---------------------------------------------------------------------------
extern "C" void kernel_launch(void* const* d_in, const int* in_sizes, int n_in,
                              void* d_out, int out_size, void* d_ws, size_t ws_size,
                              hipStream_t stream) {
  const float* enc    = (const float*)d_in[0];
  const float* w_proj = (const float*)d_in[1];
  const float* b_proj = (const float*)d_in[2];
  const float* W_ih   = (const float*)d_in[3];
  const float* b_ih   = (const float*)d_in[4];
  const float* W_hh   = (const float*)d_in[5];
  const float* b_hh   = (const float*)d_in[6];
  const float* w_out  = (const float*)d_in[7];
  const float* b_out  = (const float*)d_in[8];
  float* out = (float*)d_out;

  // workspace layout: Wb (bf16 768x256) | u[768] | c[768]
  uint16_t* Wb = (uint16_t*)d_ws;
  float* u = (float*)((char*)d_ws + (size_t)3 * HID * HID * sizeof(uint16_t));
  float* c = u + 3 * HID;

  hipLaunchKernelGGL(whh_bf16_kernel, dim3(3 * HID), dim3(HID), 0, stream, W_hh, Wb);
  hipLaunchKernelGGL(gi_precompute_kernel, dim3(3), dim3(HID), 0, stream,
                     W_ih, b_ih, w_proj, b_proj, u, c);
  hipLaunchKernelGGL(decode_kernel, dim3(NSAMP / ROWS), dim3(BLOCK), 0, stream,
                     enc, Wb, u, c, b_hh, w_out, b_out, out);
}
